// EncoderImageAggr_5712306503685
// MI455X (gfx1250) — matrix operand order repaired and checked by reading, not compiled, not run on hardware
//
#include <hip/hip_runtime.h>

// ---------------------------------------------------------------------------
// EncoderImageAggr on MI455X (gfx1250): bf16 WMMA GEMMs + fused attention
// with Tensor-Data-Mover (TDM) staging of the V tile.
// B=256, L=64, IMG_DIM=2048, D=1024, H=8, DH=128.
// ---------------------------------------------------------------------------

typedef __attribute__((ext_vector_type(16))) __bf16    v16bf;
typedef __attribute__((ext_vector_type(8)))  float     v8f;
typedef __attribute__((ext_vector_type(4)))  unsigned  v4u;
typedef __attribute__((ext_vector_type(8)))  int       v8i_t;
typedef __attribute__((ext_vector_type(4)))  int       v4i_t;

#define Bn   256
#define Ln   64
#define IMGn 2048
#define Dn   1024
#define Hn   8
#define DHn  128
#define NEGV     (-1e30f)
#define LN_EPSf  (1e-5f)
#define BN_EPSf  (1e-5f)

#if defined(__has_builtin)
#if __has_builtin(__builtin_amdgcn_tensor_load_to_lds) && __has_builtin(__builtin_amdgcn_s_wait_tensorcnt)
#define HAVE_TDM 1
#endif
#endif
#ifndef HAVE_TDM
#define HAVE_TDM 0
#endif

// f32 -> bf16, round-to-nearest-even
__device__ __forceinline__ unsigned short f2bf(float x) {
  unsigned u = __float_as_uint(x);
  u += 0x7FFFu + ((u >> 16) & 1u);
  return (unsigned short)(u >> 16);
}
__device__ __forceinline__ unsigned pack2(float lo, float hi) {
  return (unsigned)f2bf(lo) | ((unsigned)f2bf(hi) << 16);
}

union Frag { unsigned u[8]; v16bf v; };

// A fragment (16x32, 16-bit): lanes 0-15 hold M=lane with K pairs
// {0,1},{2,3},{4,5},{6,7},{16,17},{18,19},{20,21},{22,23}; lanes 16-31 shift K by +8.
__device__ __forceinline__ v16bf load_a_frag(const unsigned short* base, int stride,
                                             int row0, int k0, int lane) {
  Frag f;
  const unsigned short* p = base + (row0 + (lane & 15)) * stride + k0 + ((lane >> 4) << 3);
#pragma unroll
  for (int i = 0; i < 8; ++i) {
    int kk = 2 * i + (i >= 4 ? 8 : 0);
    f.u[i] = *(const unsigned*)(p + kk);
  }
  return f.v;
}

// B fragment (32x16, 16-bit): lanes 0-15 hold N=lane with K pairs
// {0,1},{2,3},{4,5},{6,7},{16,17},{18,19},{20,21},{22,23}; lanes 16-31 shift K by +8. Source stored as [N][K] row-major.
__device__ __forceinline__ v16bf load_b_frag(const unsigned short* base, int stride,
                                             int col0, int k0, int lane) {
  Frag f;
  const unsigned short* p = base + (col0 + (lane & 15)) * stride + k0 + ((lane >> 4) << 3);
#pragma unroll
  for (int i = 0; i < 8; ++i) f.u[i] = *(const unsigned*)(p + 2 * i + (i >= 4 ? 8 : 0));
  return f.v;
}

__device__ __forceinline__ v8f zero8() {
  v8f z = {0.f, 0.f, 0.f, 0.f, 0.f, 0.f, 0.f, 0.f};
  return z;
}

#if HAVE_TDM
// Issue one TDM 2D tile load: tile_h rows x tile_w f32 elements, row stride
// row_stride (elements), from global gsrc into LDS byte offset lds_off.
// D# built per CDNA5 ISA 8.3/8.4 (count=1, type=2 "image", data_size=4B, 2D tile).
__device__ __forceinline__ void tdm_load_2d_f32(unsigned lds_off, const float* gsrc,
                                                unsigned tile_w, unsigned tile_h,
                                                unsigned long long row_stride) {
  unsigned long long ga = (unsigned long long)(uintptr_t)gsrc;
  v4u g0;
  g0[0] = 1u;                                                // count=1, user mode
  g0[1] = lds_off;                                           // lds_addr (bytes)
  g0[2] = (unsigned)ga;                                      // global_addr[31:0]
  g0[3] = (unsigned)((ga >> 32) & 0x01FFFFFFu) | (2u << 30); // addr[56:32] | type=2

  v8i_t g1;
  g1[0] = (int)(2u << 16);                                   // data_size=2 -> 4 bytes
  g1[1] = (int)((tile_w & 0xFFFFu) << 16);                   // tensor_dim0[15:0]
  g1[2] = (int)((tile_w >> 16) | ((tile_h & 0xFFFFu) << 16));// t_dim0[31:16]|t_dim1[15:0]
  g1[3] = (int)((tile_h >> 16) | (tile_w << 16));            // t_dim1[31:16]|tile_dim0
  g1[4] = (int)(tile_h & 0xFFFFu);                           // tile_dim1 (tile_dim2=0)
  g1[5] = (int)(unsigned)(row_stride & 0xFFFFFFFFull);       // dim0_stride[31:0]
  g1[6] = (int)(unsigned)((row_stride >> 32) & 0xFFFFull);   // dim0_stride[47:32]
  g1[7] = 0;
  v4i_t gz = {0, 0, 0, 0};
#if __clang_major__ >= 23
  v8i_t gz8 = {0, 0, 0, 0, 0, 0, 0, 0};
  __builtin_amdgcn_tensor_load_to_lds(g0, g1, gz, gz, gz8, 0);
#else
  __builtin_amdgcn_tensor_load_to_lds(g0, g1, gz, gz, 0);
#endif
}
#endif

// ---------------------------------------------------------------------------
// C[M,N] = act( A[M,K] @ W[N,K]^T + bias )  with MODE: 0=bias, 1=+ReLU, 2=+BN+ReLU.
// Block tile 128x64, 8 waves (4x2), each wave computes 2x2 16x16 WMMA tiles.
// f32 loaded from HBM, converted to bf16 while staging into LDS.
// ---------------------------------------------------------------------------
template <int MODE>
__global__ __launch_bounds__(256) void gemm_bf16_wmma(
    const float* __restrict__ A, const float* __restrict__ W,
    const float* __restrict__ bias,
    const float* __restrict__ bng, const float* __restrict__ bnb,
    const float* __restrict__ bnm, const float* __restrict__ bnv,
    float* __restrict__ C, int M, int N, int K) {
  __shared__ alignas(16) unsigned short As[128][34];  // +2 pad: stride 68B, conflict-free
  __shared__ alignas(16) unsigned short Bs[64][34];

  const int tid  = threadIdx.x;
  const int lane = tid & 31;
  const int wave = tid >> 5;
  const int wm = wave & 3;   // 4 wave rows * 32
  const int wn = wave >> 2;  // 2 wave cols * 32
  const int m0 = blockIdx.x * 128;
  const int n0 = blockIdx.y * 64;

  v8f acc[2][2];
  acc[0][0] = zero8(); acc[0][1] = zero8();
  acc[1][0] = zero8(); acc[1][1] = zero8();

  for (int k0 = 0; k0 < K; k0 += 32) {
    // Stage A tile 128x32 (1024 float4 slots / 256 threads)
    for (int s = tid; s < 1024; s += 256) {
      int r = s >> 3, c = (s & 7) << 2;
      const float* ap = A + (size_t)(m0 + r) * K + k0 + c;
      float4 f = *(const float4*)ap;
      if (k0 + 32 < K) __builtin_prefetch(ap + 32, 0, 1);  // global_prefetch_b8
      *(unsigned*)&As[r][c]     = pack2(f.x, f.y);
      *(unsigned*)&As[r][c + 2] = pack2(f.z, f.w);
    }
    // Stage W tile 64x32
    for (int s = tid; s < 512; s += 256) {
      int r = s >> 3, c = (s & 7) << 2;
      const float* wp = W + (size_t)(n0 + r) * K + k0 + c;
      float4 f = *(const float4*)wp;
      *(unsigned*)&Bs[r][c]     = pack2(f.x, f.y);
      *(unsigned*)&Bs[r][c + 2] = pack2(f.z, f.w);
    }
    __syncthreads();

#pragma unroll
    for (int tm = 0; tm < 2; ++tm) {
      v16bf af = load_a_frag(&As[0][0], 34, wm * 32 + tm * 16, 0, lane);
#pragma unroll
      for (int tn = 0; tn < 2; ++tn) {
        v16bf bf = load_b_frag(&Bs[0][0], 34, wn * 32 + tn * 16, 0, lane);
        acc[tm][tn] = __builtin_amdgcn_wmma_f32_16x16x32_bf16(
            false, af, false, bf, (short)0, acc[tm][tn], false, false);
      }
    }
    __syncthreads();
  }

  // Epilogue: C/D layout -> lanes 0-15 N=lane, VGPR r -> M=r (lanes16-31: M=8+r)
#pragma unroll
  for (int tm = 0; tm < 2; ++tm) {
#pragma unroll
    for (int tn = 0; tn < 2; ++tn) {
      int col = n0 + wn * 32 + tn * 16 + (lane & 15);
      float bv = bias[col];
      float sc = 1.f, sh = 0.f;
      if (MODE == 2) {
        sc = bng[col] * rsqrtf(bnv[col] + BN_EPSf);
        sh = bnb[col] - bnm[col] * sc;
      }
#pragma unroll
      for (int r = 0; r < 8; ++r) {
        int row = m0 + wm * 32 + tm * 16 + ((lane >> 4) << 3) + r;
        float v = acc[tm][tn][r] + bv;
        if (MODE == 2) { v = v * sc + sh; v = fmaxf(v, 0.f); }
        if (MODE == 1) { v = fmaxf(v, 0.f); }
        C[(size_t)row * N + col] = v;
      }
    }
  }
}

// ---------------------------------------------------------------------------
// Fused attention per (b, head): S = QK^T/sqrt(DH) + mask; softmax; ctx = P V.
// WMMA for both matmuls; V tile staged into LDS by the Tensor Data Mover.
// LDS overlays (58368 B total):
//   [0,33280)      Qs/Ks bf16 (phase 1)   -> later Vf f32 [0,32768) (TDM dest)
//   [32768,49664)  Vt bf16 (phase 2)      overlays dead Sc
//   [33280,49920)  Sc f32 scores          dead after softmax
//   [49920,58368)  Ps bf16 probabilities
// ---------------------------------------------------------------------------
__global__ __launch_bounds__(256) void attention_kernel(
    const float* __restrict__ qkv, const int* __restrict__ lengths,
    float* __restrict__ ctx) {
  __shared__ alignas(16) unsigned char smem[58368];
  unsigned short* Qs = (unsigned short*)smem;               // [64][130] bf16
  unsigned short* Ks = Qs + 64 * 130;                       // [64][130] bf16
  float*          Vf = (float*)smem;                        // [64][128] f32 (TDM dest)
  unsigned short* Vt = (unsigned short*)(smem + 32768);     // [128][66] bf16
  float*          Sc = (float*)(smem + 33280);              // [64][65]  f32
  unsigned short* Ps = (unsigned short*)(smem + 49920);     // [64][66]  bf16

  const int tid  = threadIdx.x;
  const int lane = tid & 31;
  const int wave = tid >> 5;
  const int b  = blockIdx.x >> 3;
  const int hh = blockIdx.x & 7;
  const int len = lengths[b];
  const size_t rowbase = (size_t)b * Ln * (3 * Dn);

  // Stage Q and K as bf16
  for (int s = tid; s < 2048; s += 256) {
    int l = s >> 5, dv = (s & 31) << 2;
    float4 q4 = *(const float4*)(qkv + rowbase + (size_t)l * (3 * Dn) + hh * DHn + dv);
    float4 k4 = *(const float4*)(qkv + rowbase + (size_t)l * (3 * Dn) + Dn + hh * DHn + dv);
    *(unsigned*)&Qs[l * 130 + dv]     = pack2(q4.x, q4.y);
    *(unsigned*)&Qs[l * 130 + dv + 2] = pack2(q4.z, q4.w);
    *(unsigned*)&Ks[l * 130 + dv]     = pack2(k4.x, k4.y);
    *(unsigned*)&Ks[l * 130 + dv + 2] = pack2(k4.z, k4.w);
  }
  __syncthreads();

  // S = Q K^T * scale + key mask : 16 tiles of 16x16, 2 per wave
  const float scale = 0.08838834764831845f;  // 1/sqrt(128)
#pragma unroll
  for (int t = 0; t < 2; ++t) {
    int tile = wave * 2 + t;
    int ti = tile >> 2, tj = tile & 3;
    v8f acc = zero8();
#pragma unroll
    for (int k0 = 0; k0 < 128; k0 += 32) {
      v16bf af = load_a_frag(Qs, 130, ti * 16, k0, lane);
      v16bf bf = load_b_frag(Ks, 130, tj * 16, k0, lane);
      acc = __builtin_amdgcn_wmma_f32_16x16x32_bf16(false, af, false, bf, (short)0,
                                                    acc, false, false);
    }
    int col = tj * 16 + (lane & 15);
    float mask = (col < len) ? 0.f : NEGV;
#pragma unroll
    for (int r = 0; r < 8; ++r) {
      int row = ti * 16 + ((lane >> 4) << 3) + r;
      Sc[row * 65 + col] = acc[r] * scale + mask;
    }
  }
  __syncthreads();

  // Row softmax (all 64 query rows; keys masked above), write P as bf16
  if (tid < 64) {
    int row = tid;
    float m = -3.0e38f;
    for (int c = 0; c < 64; ++c) m = fmaxf(m, Sc[row * 65 + c]);
    float s = 0.f;
    for (int c = 0; c < 64; ++c) s += __expf(Sc[row * 65 + c] - m);
    float inv = 1.f / s;
    for (int c = 0; c < 64; ++c)
      Ps[row * 66 + c] = f2bf(__expf(Sc[row * 65 + c] - m) * inv);
  }
  __syncthreads();

  // Stage V tile (64 rows x 128 f32, row stride 3072) into LDS.
#if HAVE_TDM
  if (tid < 32) {  // one wave issues the DMA; TDM ignores EXEC
    tdm_load_2d_f32((unsigned)(uintptr_t)(void*)Vf,
                    qkv + rowbase + 2 * Dn + hh * DHn,
                    DHn, Ln, (unsigned long long)(3 * Dn));
    __builtin_amdgcn_s_wait_tensorcnt(0);
  }
  __syncthreads();
#else
  for (int s = tid; s < 2048; s += 256) {
    int l = s >> 5, dv = (s & 31) << 2;
    float4 v4 = *(const float4*)(qkv + rowbase + (size_t)l * (3 * Dn) + 2 * Dn + hh * DHn + dv);
    *(float4*)&Vf[l * DHn + dv] = v4;
  }
  __syncthreads();
#endif

  // Transpose-convert to bf16: Vt[d][l]
  for (int s = tid; s < 2048; s += 256) {
    int l = s >> 5, dv = (s & 31) << 2;
    float4 v4 = *(const float4*)&Vf[l * DHn + dv];
    Vt[(dv + 0) * 66 + l] = f2bf(v4.x);
    Vt[(dv + 1) * 66 + l] = f2bf(v4.y);
    Vt[(dv + 2) * 66 + l] = f2bf(v4.z);
    Vt[(dv + 3) * 66 + l] = f2bf(v4.w);
  }
  __syncthreads();

  // ctx = P @ V : M=64, N=128, K=64 -> 32 tiles, 4 per wave
#pragma unroll
  for (int t = 0; t < 4; ++t) {
    int tile = wave * 4 + t;
    int ti = tile >> 3, tj = tile & 7;
    v8f acc = zero8();
#pragma unroll
    for (int k0 = 0; k0 < 64; k0 += 32) {
      v16bf af = load_a_frag(Ps, 66, ti * 16, k0, lane);
      v16bf bf = load_b_frag(Vt, 66, tj * 16, k0, lane);
      acc = __builtin_amdgcn_wmma_f32_16x16x32_bf16(false, af, false, bf, (short)0,
                                                    acc, false, false);
    }
    int d = tj * 16 + (lane & 15);
#pragma unroll
    for (int r = 0; r < 8; ++r) {
      int q = ti * 16 + ((lane >> 4) << 3) + r;
      ctx[(size_t)(b * Ln + q) * Dn + hh * DHn + d] = acc[r];
    }
  }
}

// ---------------------------------------------------------------------------
// Elementwise / reduction kernels
// ---------------------------------------------------------------------------
__device__ __forceinline__ float block_sum(float v, float* red, int tid) {
  red[tid] = v;
  __syncthreads();
  for (int s = 128; s > 0; s >>= 1) {
    if (tid < s) red[tid] += red[tid + s];
    __syncthreads();
  }
  float r = red[0];
  __syncthreads();
  return r;
}

// y = LayerNorm(a + c) * g + beta, one block per token row (D=1024)
__global__ __launch_bounds__(256) void add_ln_kernel(
    const float* __restrict__ a, const float* __restrict__ c,
    const float* __restrict__ g, const float* __restrict__ beta,
    float* __restrict__ y) {
  __shared__ float red[256];
  const int row = blockIdx.x, tid = threadIdx.x;
  const size_t base = (size_t)row * Dn;
  float v[4];
  float s = 0.f;
#pragma unroll
  for (int i = 0; i < 4; ++i) {
    int col = tid + i * 256;
    v[i] = a[base + col] + c[base + col];
    s += v[i];
  }
  float mean = block_sum(s, red, tid) * (1.f / Dn);
  float q = 0.f;
#pragma unroll
  for (int i = 0; i < 4; ++i) {
    float d = v[i] - mean;
    q += d * d;
  }
  float var = block_sum(q, red, tid) * (1.f / Dn);
  float inv = rsqrtf(var + LN_EPSf);
#pragma unroll
  for (int i = 0; i < 4; ++i) {
    int col = tid + i * 256;
    y[base + col] = (v[i] - mean) * inv * g[col] + beta[col];
  }
}

// mean of top-2 over valid tokens, per (b, d)
__global__ __launch_bounds__(256) void topk_kernel(
    const float* __restrict__ emb, const int* __restrict__ lengths,
    float* __restrict__ res) {
  int idx = blockIdx.x * 256 + threadIdx.x;  // b*D + d
  int b = idx >> 10, d = idx & 1023;
  int len = lengths[b];  // len >= 2 guaranteed
  float m1 = NEGV, m2 = NEGV;
  for (int l = 0; l < len; ++l) {
    float v = emb[(size_t)(b * Ln + l) * Dn + d];
    if (v > m1) { m2 = m1; m1 = v; }
    else if (v > m2) { m2 = v; }
  }
  res[idx] = 0.5f * (m1 + m2);
}

// avg-pool over valid tokens, mix with emb_res, L2-normalize per row
__global__ __launch_bounds__(256) void finalize_kernel(
    const float* __restrict__ x, const float* __restrict__ res,
    const int* __restrict__ lengths, float* __restrict__ out) {
  __shared__ float red[256];
  const int b = blockIdx.x, tid = threadIdx.x;
  const int len = lengths[b];
  const float invlen = 1.f / (float)len;
  float o[4];
  float ss = 0.f;
#pragma unroll
  for (int i = 0; i < 4; ++i) {
    int d = tid + i * 256;
    float s = 0.f;
    for (int l = 0; l < len; ++l) s += x[(size_t)(b * Ln + l) * Dn + d];
    float v = 0.5f * res[b * Dn + d] + 0.5f * (s * invlen);
    o[i] = v;
    ss += v * v;
  }
  float tot = block_sum(ss, red, tid);
  float inv = 1.f / (sqrtf(tot) + 1e-8f);
#pragma unroll
  for (int i = 0; i < 4; ++i) out[(size_t)b * Dn + tid + i * 256] = o[i] * inv;
}

// ---------------------------------------------------------------------------
// Host-side orchestration
// ---------------------------------------------------------------------------
extern "C" void kernel_launch(void* const* d_in, const int* in_sizes, int n_in,
                              void* d_out, int out_size, void* d_ws, size_t ws_size,
                              hipStream_t stream) {
  const float* images  = (const float*)d_in[0];
  const int*   lengths = (const int*)d_in[1];
  const float* fc1_w = (const float*)d_in[2];
  const float* fc1_b = (const float*)d_in[3];
  const float* bn_g  = (const float*)d_in[4];
  const float* bn_b  = (const float*)d_in[5];
  const float* bn_m  = (const float*)d_in[6];
  const float* bn_v  = (const float*)d_in[7];
  const float* fc2_w = (const float*)d_in[8];
  const float* fc2_b = (const float*)d_in[9];
  const float* in_w  = (const float*)d_in[10];
  const float* in_b  = (const float*)d_in[11];
  const float* out_w = (const float*)d_in[12];
  const float* out_b = (const float*)d_in[13];
  const float* ln1_g = (const float*)d_in[14];
  const float* ln1_b = (const float*)d_in[15];
  const float* ff1_w = (const float*)d_in[16];
  const float* ff1_b = (const float*)d_in[17];
  const float* ff2_w = (const float*)d_in[18];
  const float* ff2_b = (const float*)d_in[19];
  const float* ln2_g = (const float*)d_in[20];
  const float* ln2_b = (const float*)d_in[21];

  const int M = Bn * Ln;  // 16384 token rows
  float* ws = (float*)d_ws;
  // Workspace layout (floats), with reuse:
  float* emb     = ws;                                   // 16384*1024 (persists: residual + topk)
  float* qkvbuf  = emb + (size_t)M * Dn;                 // 16384*3072
  float* xbuf    = qkvbuf + (size_t)M * 3 * Dn;          // 16384*1024 (ctx -> f1 -> x2)
  float* x1      = xbuf + (size_t)M * Dn;                // 16384*1024
  float* hbuf    = x1 + (size_t)M * Dn;                  // 16384*512
  float* emb_res = hbuf + (size_t)M * 512;               // 256*1024
  float* ctxp = qkvbuf;                                  // reuse (qkv dead after attention)
  float* f2   = qkvbuf + (size_t)M * Dn;                 // reuse (disjoint from ctxp)
  float* ctx  = xbuf;
  float* f1   = xbuf;
  float* x2   = xbuf;

  // 1) h = ReLU(BN(images @ fc1^T + b))         [16384 x 512, K=2048]
  gemm_bf16_wmma<2><<<dim3(M / 128, 512 / 64), 256, 0, stream>>>(
      images, fc1_w, fc1_b, bn_g, bn_b, bn_m, bn_v, hbuf, M, 512, IMGn);
  // 2) emb = h @ fc2^T + b                      [16384 x 1024, K=512]
  gemm_bf16_wmma<0><<<dim3(M / 128, Dn / 64), 256, 0, stream>>>(
      hbuf, fc2_w, fc2_b, nullptr, nullptr, nullptr, nullptr, emb, M, Dn, 512);
  // 3) qkv = emb @ in_w^T + in_b                [16384 x 3072, K=1024]
  gemm_bf16_wmma<0><<<dim3(M / 128, 3 * Dn / 64), 256, 0, stream>>>(
      emb, in_w, in_b, nullptr, nullptr, nullptr, nullptr, qkvbuf, M, 3 * Dn, Dn);
  // 4) attention -> ctx                         [per (b, head)]
  attention_kernel<<<Bn * Hn, 256, 0, stream>>>(qkvbuf, lengths, ctx);
  // 5) ctxp = ctx @ out_w^T + out_b
  gemm_bf16_wmma<0><<<dim3(M / 128, Dn / 64), 256, 0, stream>>>(
      ctx, out_w, out_b, nullptr, nullptr, nullptr, nullptr, ctxp, M, Dn, Dn);
  // 6) x1 = LN(emb + ctxp)
  add_ln_kernel<<<M, 256, 0, stream>>>(emb, ctxp, ln1_g, ln1_b, x1);
  // 7) f1 = ReLU(x1 @ ff1^T + b)
  gemm_bf16_wmma<1><<<dim3(M / 128, Dn / 64), 256, 0, stream>>>(
      x1, ff1_w, ff1_b, nullptr, nullptr, nullptr, nullptr, f1, M, Dn, Dn);
  // 8) f2 = f1 @ ff2^T + b
  gemm_bf16_wmma<0><<<dim3(M / 128, Dn / 64), 256, 0, stream>>>(
      f1, ff2_w, ff2_b, nullptr, nullptr, nullptr, nullptr, f2, M, Dn, Dn);
  // 9) x2 = LN(x1 + f2)
  add_ln_kernel<<<M, 256, 0, stream>>>(x1, f2, ln2_g, ln2_b, x2);
  // 10) emb_res = mean(top-2 over valid tokens of emb)
  topk_kernel<<<Bn * Dn / 256, 256, 0, stream>>>(emb, lengths, emb_res);
  // 11) out = normalize(0.5*emb_res + 0.5*avg_pool(x2))
  finalize_kernel<<<Bn, 256, 0, stream>>>(x2, emb_res, lengths, (float*)d_out);

  (void)in_sizes; (void)n_in; (void)out_size; (void)ws_size;
}
